// SelfAttentionModel_996432413165
// MI455X (gfx1250) — matrix operand order repaired and checked by reading, not compiled, not run on hardware
//
#include <hip/hip_runtime.h>

// ---------------------------------------------------------------------------
// Shapes (fixed by the reference)
// ---------------------------------------------------------------------------
#define S_  2048
#define B_  4
#define D_  256     // hidden_dim
#define E_  256     // embed
#define H_  8
#define HD_ 32      // head dim
#define REQ_ 16     // band half-width: keys j with |i-j| < 16  -> 31 keys
#define M_  (S_ * B_)          // 8192 token rows

typedef __attribute__((ext_vector_type(16))) __bf16        v16bf;
typedef __attribute__((ext_vector_type(8)))  float         v8f;
typedef __attribute__((ext_vector_type(4))) unsigned int   v4u;

// exact parameter types for the gfx1250 async LDS-copy builtin
// (from the compiler diagnostic: "__attribute__((__vector_size__(4*sizeof(int)))) int __device__ *")
typedef int gvec4i __attribute__((vector_size(16)));
typedef __attribute__((address_space(1))) gvec4i* as1_v4i;
typedef __attribute__((address_space(3))) gvec4i* as3_v4i;

union FragU { v16bf bf; v4u u[2]; };

__device__ __forceinline__ unsigned short f2bf(float f) {
    union { float f; unsigned int u; } c; c.f = f;
    unsigned int u = c.u;
    unsigned int r = u + 0x7FFFu + ((u >> 16) & 1u);           // RNE
    if ((u & 0x7F800000u) == 0x7F800000u) r = u;               // inf/nan
    return (unsigned short)(r >> 16);
}

__device__ __forceinline__ void wait_async0() {
#if __has_builtin(__builtin_amdgcn_s_wait_asynccnt)
    __builtin_amdgcn_s_wait_asynccnt(0);
#else
    asm volatile("s_wait_asynccnt 0" ::: "memory");
#endif
}

// ---------------------------------------------------------------------------
// Kernel 1: fold the back-to-back linear layers.
//   Wc  (768x256) = Win[part] @ W{q,k,v},  bc = Win[part]@b{q,k,v} + bin
//   Wof (256x256) = Wfc @ Wout,            bof = Wfc@bout + bfc
// Tiny (4 x 16.7 MFLOP) -> naive. Stored as bf16 for the WMMA passes.
// ---------------------------------------------------------------------------
__global__ void combine_weights(const float* __restrict__ Wq, const float* __restrict__ bq,
                                const float* __restrict__ Wk, const float* __restrict__ bk,
                                const float* __restrict__ Wv, const float* __restrict__ bv,
                                const float* __restrict__ Win, const float* __restrict__ bin,
                                const float* __restrict__ Wout, const float* __restrict__ bout,
                                const float* __restrict__ Wfc, const float* __restrict__ bfc,
                                unsigned short* __restrict__ Wc, float* __restrict__ bc,
                                unsigned short* __restrict__ Wof, float* __restrict__ bof) {
    int g = blockIdx.x;            // 0..1023
    int t = threadIdx.x;           // 0..255 output column
    if (g < 3 * E_) {              // combined QKV weight row g
        int part = g >> 8;         // 0=q 1=k 2=v
        const float* Wx = (part == 0) ? Wq : (part == 1) ? Wk : Wv;
        const float* winrow = Win + (size_t)g * E_;
        float acc = 0.f;
        for (int j = 0; j < E_; ++j) acc += winrow[j] * Wx[(size_t)j * D_ + t];
        Wc[(size_t)g * D_ + t] = f2bf(acc);
        if (t == 0) {
            const float* bx = (part == 0) ? bq : (part == 1) ? bk : bv;
            float bb = 0.f;
            for (int j = 0; j < E_; ++j) bb += winrow[j] * bx[j];
            bc[g] = bb + bin[g];
        }
    } else {                       // Wof row i
        int i = g - 3 * E_;
        const float* frow = Wfc + (size_t)i * E_;
        float acc = 0.f;
        for (int u = 0; u < E_; ++u) acc += frow[u] * Wout[(size_t)u * E_ + t];
        Wof[(size_t)i * E_ + t] = f2bf(acc);
        if (t == 0) {
            float bb = 0.f;
            for (int u = 0; u < E_; ++u) bb += frow[u] * bout[u];
            bof[i] = bb + bfc[i];
        }
    }
}

// ---------------------------------------------------------------------------
// Kernel 2: fp32 -> bf16 activation convert
// ---------------------------------------------------------------------------
__global__ void cvt_bf16(const float* __restrict__ x, unsigned short* __restrict__ y, int n) {
    int i = blockIdx.x * blockDim.x + threadIdx.x;
    if (i < n) y[i] = f2bf(x[i]);
}

// ---------------------------------------------------------------------------
// Fragment loaders (wave32, contiguous-k order this code uses for v_wmma_f32_16x16x32_bf16, same for A and B)
// A: row = lane&15; elems 0..15 -> K=kb..kb+15; kb=16*(lane>=16)
// B (reading rows of W for X@W^T): col = lane&15; elems 0..15 -> K = 16*(lane>=16)..+16
// ---------------------------------------------------------------------------
__device__ __forceinline__ v16bf load_fragA(const unsigned short* rowPtr /*row base + kb*/,
                                            int k0) {
    FragU f;
    const v4u* p = (const v4u*)(rowPtr + k0);
    f.u[0] = p[0];       // K = k0+kb .. +8
    f.u[1] = p[1];       // K = k0+kb+8 .. +8
    return f.bf;
}
__device__ __forceinline__ v16bf load_fragB(const unsigned short* base, int wrow, int k0,
                                            int kbB) {
    FragU f;
    const v4u* p = (const v4u*)(base + (size_t)wrow * 256 + k0 + kbB);
    f.u[0] = p[0];       // K = k0+kbB .. +8
    f.u[1] = p[1];       // K = k0+kbB+8 .. +8
    return f.bf;
}

// ---------------------------------------------------------------------------
// Kernel 3: QKV GEMM  (8192 x 768 x 256, bf16 in / f32 out)
// Block = 128(M) x 64(N); 8 waves, each a 16x64 strip. The 64x256 bf16 weight
// strip (32 KB) is staged into LDS ONCE per block with the gfx1250 async
// LDS-copy engine (ASYNCcnt), then B fragments come from ds_load_b128 while
// A fragments stream from global.
// ---------------------------------------------------------------------------
__global__ void qkv_gemm(const unsigned short* __restrict__ Xbf,
                         const unsigned short* __restrict__ Wc,
                         const float* __restrict__ bc,
                         float* __restrict__ QKV) {
    __shared__ unsigned short Bsh[64 * 256];     // 32 KB weight strip
    const int tid  = threadIdx.x;
    const int lane = tid & 31;
    const int wav  = tid >> 5;                   // 0..7

    const int nstrip = blockIdx.x % 12;          // 768 / 64
    const int mblk   = blockIdx.x / 12;          // 0..63 (128 rows each)
    const int col0   = nstrip * 64;
    const int row0   = mblk * 128 + wav * 16;

    // ---- stage weight strip Wc[col0 .. col0+64) x 256 into LDS ----
    {
        const unsigned short* src = Wc + (size_t)col0 * 256;   // 2048 x 16B chunks
#if __has_builtin(__builtin_amdgcn_global_load_async_to_lds_b128)
#pragma unroll
        for (int c = 0; c < 8; ++c) {
            const int chunk = c * 256 + tid;                   // coalesced per pass
            __builtin_amdgcn_global_load_async_to_lds_b128(
                (as1_v4i)(src + (size_t)chunk * 8),
                (as3_v4i)(Bsh + (size_t)chunk * 8),
                0, 0);
        }
        wait_async0();
#else
        const v4u* s4 = (const v4u*)src;
        v4u* d4 = (v4u*)Bsh;
        for (int c = tid; c < 2048; c += 256) d4[c] = s4[c];
#endif
        __syncthreads();
    }

    const int r   = lane & 15;
    const int kb  = (lane >> 4) * 16;
    const int kbB = (lane >> 4) * 16;
    const unsigned short* arow = Xbf + (size_t)(row0 + r) * 256 + kb;

    v8f acc0 = {}, acc1 = {}, acc2 = {}, acc3 = {};
    for (int k0 = 0; k0 < 256; k0 += 32) {
        v16bf a  = load_fragA(arow, k0);
        v16bf b0 = load_fragB(Bsh,  0 + r, k0, kbB);           // ds_load_b128 x2
        v16bf b1 = load_fragB(Bsh, 16 + r, k0, kbB);
        v16bf b2 = load_fragB(Bsh, 32 + r, k0, kbB);
        v16bf b3 = load_fragB(Bsh, 48 + r, k0, kbB);
        acc0 = __builtin_amdgcn_wmma_f32_16x16x32_bf16(false, a, false, b0, (short)0, acc0, false, false);
        acc1 = __builtin_amdgcn_wmma_f32_16x16x32_bf16(false, a, false, b1, (short)0, acc1, false, false);
        acc2 = __builtin_amdgcn_wmma_f32_16x16x32_bf16(false, a, false, b2, (short)0, acc2, false, false);
        acc3 = __builtin_amdgcn_wmma_f32_16x16x32_bf16(false, a, false, b3, (short)0, acc3, false, false);
    }
    const int mofs = (lane >> 4) * 8;
    const int nlo  = lane & 15;
#pragma unroll
    for (int rr = 0; rr < 8; ++rr) {
        const size_t m = (size_t)(row0 + rr + mofs);
        int n0 = col0 + nlo;
        QKV[m * 768 + n0     ] = acc0[rr] + bc[n0     ];
        QKV[m * 768 + n0 + 16] = acc1[rr] + bc[n0 + 16];
        QKV[m * 768 + n0 + 32] = acc2[rr] + bc[n0 + 32];
        QKV[m * 768 + n0 + 48] = acc3[rr] + bc[n0 + 48];
    }
}

// ---------------------------------------------------------------------------
// Kernel 4: banded attention, one wave per (s, b, h) query row.
// 31 keys (|i-j|<16), hd=32 == wave32 lanes. fp32 + shuffles for accuracy.
// Output ctx directly in bf16 for the next WMMA pass.
// ---------------------------------------------------------------------------
__global__ void attn_band(const float* __restrict__ QKV, unsigned short* __restrict__ CTX) {
    const int lane = threadIdx.x & 31;
    const int wid  = blockIdx.x * (blockDim.x >> 5) + (threadIdx.x >> 5);
    const int h  = wid & (H_ - 1);
    const int sb = wid >> 3;                     // s*B + b
    const int b  = sb & (B_ - 1);
    const int s  = sb >> 2;

    const float q = QKV[(size_t)sb * 768 + h * HD_ + lane];  // lane d holds q[d]
    const float scale = 0.17677669529663687f;                // 1/sqrt(32)

    float score = -1e30f;                                    // lane j holds score_j
#pragma unroll
    for (int j = 0; j < 2 * REQ_ - 1; ++j) {
        const int key = s - (REQ_ - 1) + j;
        float dot = 0.f;
        if (key >= 0 && key < S_)
            dot = q * QKV[((size_t)key * B_ + b) * 768 + 256 + h * HD_ + lane];
        for (int o = 16; o >= 1; o >>= 1) dot += __shfl_xor(dot, o, 32);
        if (lane == j && key >= 0 && key < S_) score = dot * scale;
    }
    // softmax over the band (masked lanes -> weight 0)
    float mx = score;
    for (int o = 16; o >= 1; o >>= 1) mx = fmaxf(mx, __shfl_xor(mx, o, 32));
    float w = __expf(score - mx);
    float sum = w;
    for (int o = 16; o >= 1; o >>= 1) sum += __shfl_xor(sum, o, 32);
    w /= sum;

    // ctx[d] = sum_j w_j * V[key_j][d]; lane d accumulates, V reads coalesced
    float acc = 0.f;
#pragma unroll
    for (int j = 0; j < 2 * REQ_ - 1; ++j) {
        const int key = s - (REQ_ - 1) + j;
        const float wj = __shfl(w, j, 32);
        if (key >= 0 && key < S_)
            acc += wj * QKV[((size_t)key * B_ + b) * 768 + 512 + h * HD_ + lane];
    }
    CTX[(size_t)sb * E_ + h * HD_ + lane] = f2bf(acc);
}

// ---------------------------------------------------------------------------
// Kernel 5: output GEMM (ctx @ Wof^T + bof) + residual + LayerNorm, fused.
// Block = 16 rows x 256 cols; 8 waves, each wave a 16x32 slab (2 WMMA accums).
// y tile staged in LDS for the row-wise LayerNorm.
// ---------------------------------------------------------------------------
__global__ void out_gemm_ln(const unsigned short* __restrict__ CTXbf,
                            const unsigned short* __restrict__ Wof,
                            const float* __restrict__ bof,
                            const float* __restrict__ mod,
                            const float* __restrict__ gamma,
                            const float* __restrict__ beta,
                            float* __restrict__ out) {
    __shared__ float ytile[16 * 256];
    const int lane = threadIdx.x & 31;
    const int w    = threadIdx.x >> 5;           // 0..7
    const int row0 = blockIdx.x * 16;
    const int col0 = w * 32;

    const int r   = lane & 15;
    const int kb  = (lane >> 4) * 16;
    const int kbB = (lane >> 4) * 16;
    const unsigned short* arow = CTXbf + (size_t)(row0 + r) * 256 + kb;

    v8f acc0 = {}, acc1 = {};
    for (int k0 = 0; k0 < 256; k0 += 32) {
        v16bf a  = load_fragA(arow, k0);
        v16bf b0 = load_fragB(Wof, col0 +  0 + r, k0, kbB);
        v16bf b1 = load_fragB(Wof, col0 + 16 + r, k0, kbB);
        acc0 = __builtin_amdgcn_wmma_f32_16x16x32_bf16(false, a, false, b0, (short)0, acc0, false, false);
        acc1 = __builtin_amdgcn_wmma_f32_16x16x32_bf16(false, a, false, b1, (short)0, acc1, false, false);
    }
    const int mofs = (lane >> 4) * 8;
    const int nlo  = lane & 15;
#pragma unroll
    for (int rr = 0; rr < 8; ++rr) {
        const int lm = rr + mofs;
        const size_t m = (size_t)(row0 + lm);
        const int n0 = col0 + nlo, n1 = col0 + 16 + nlo;
        ytile[lm * 256 + n0] = acc0[rr] + bof[n0] + mod[m * 256 + n0];
        ytile[lm * 256 + n1] = acc1[rr] + bof[n1] + mod[m * 256 + n1];
    }
    __syncthreads();

    // LayerNorm: wave w normalizes rows 2w and 2w+1
#pragma unroll
    for (int rr = 0; rr < 2; ++rr) {
        const int lr = 2 * w + rr;
        float s1 = 0.f, s2 = 0.f;
        for (int c = lane; c < 256; c += 32) {
            float v = ytile[lr * 256 + c];
            s1 += v; s2 += v * v;
        }
        for (int o = 16; o >= 1; o >>= 1) {
            s1 += __shfl_xor(s1, o, 32);
            s2 += __shfl_xor(s2, o, 32);
        }
        const float mu   = s1 * (1.f / 256.f);
        const float var  = s2 * (1.f / 256.f) - mu * mu;
        const float rstd = rsqrtf(var + 1e-6f);
        for (int c = lane; c < 256; c += 32) {
            out[(size_t)(row0 + lr) * 256 + c] =
                (ytile[lr * 256 + c] - mu) * rstd * gamma[c] + beta[c];
        }
    }
}

// ---------------------------------------------------------------------------
// Host launcher
// ---------------------------------------------------------------------------
extern "C" void kernel_launch(void* const* d_in, const int* in_sizes, int n_in,
                              void* d_out, int out_size, void* d_ws, size_t ws_size,
                              hipStream_t stream) {
    const float* mod   = (const float*)d_in[0];
    // d_in[1] = mask (unused by the reference branch)
    const float* Wq  = (const float*)d_in[2];  const float* bq  = (const float*)d_in[3];
    const float* Wk  = (const float*)d_in[4];  const float* bk  = (const float*)d_in[5];
    const float* Wv  = (const float*)d_in[6];  const float* bv  = (const float*)d_in[7];
    const float* Win = (const float*)d_in[8];  const float* bin = (const float*)d_in[9];
    const float* Wout= (const float*)d_in[10]; const float* bout= (const float*)d_in[11];
    const float* Wfc = (const float*)d_in[12]; const float* bfc = (const float*)d_in[13];
    const float* gamma = (const float*)d_in[14];
    const float* beta  = (const float*)d_in[15];
    float* out = (float*)d_out;

    // workspace carve-out (256B aligned)
    char* ws = (char*)d_ws;
    size_t off = 0;
    auto take = [&](size_t bytes) { void* p = ws + off; off = (off + bytes + 255) & ~(size_t)255; return p; };
    unsigned short* Xbf   = (unsigned short*)take((size_t)M_ * D_ * 2);        //  4 MB
    unsigned short* Wc    = (unsigned short*)take((size_t)3 * E_ * D_ * 2);    //  384 KB
    float*          bc    = (float*)take((size_t)3 * E_ * 4);
    unsigned short* WofBf = (unsigned short*)take((size_t)D_ * E_ * 2);        //  128 KB
    float*          bof   = (float*)take((size_t)D_ * 4);
    float*          QKV   = (float*)take((size_t)M_ * 3 * E_ * 4);             //  24 MB
    unsigned short* CTX   = (unsigned short*)take((size_t)M_ * E_ * 2);        //  4 MB

    // 1) fold linear layers
    combine_weights<<<4 * 256, 256, 0, stream>>>(Wq, bq, Wk, bk, Wv, bv, Win, bin,
                                                 Wout, bout, Wfc, bfc, Wc, bc, WofBf, bof);
    // 2) activations -> bf16
    cvt_bf16<<<(M_ * D_ + 255) / 256, 256, 0, stream>>>(mod, Xbf, M_ * D_);
    // 3) QKV projection: 8192x768x256, LDS-staged weights, 16x64 strip per wave
    qkv_gemm<<<64 * 12, 256, 0, stream>>>(Xbf, Wc, bc, QKV);
    // 4) banded attention: one wave per (s,b,h)
    attn_band<<<(M_ * H_) / 8, 256, 0, stream>>>(QKV, CTX);
    // 5) out-proj + residual + LayerNorm
    out_gemm_ln<<<M_ / 16, 256, 0, stream>>>(CTX, WofBf, bof, mod, gamma, beta, out);

    (void)in_sizes; (void)n_in; (void)out_size; (void)ws_size;
}
